// MHAwithKVcache_52536039965332
// MI455X (gfx1250) — matrix operand order repaired and checked by reading, not compiled, not run on hardware
//

#include <hip/hip_runtime.h>
#include <hip/hip_bf16.h>

// ---------------------------------------------------------------------------
// MHA with KV cache on gfx1250 (MI455X).
// fp32 -> bf16 once; GEMMs + flash attention on v_wmma_f32_16x16x32_bf16
// (wave32), fp32 accumulation. Tiles staged global->LDS with
// GLOBAL_LOAD_ASYNC_TO_LDS_B128 (ASYNCcnt) into double-buffered LDS.
// ---------------------------------------------------------------------------

#define HIDDEN_   2048
#define HEADS_    16
#define D_        128      // head dim
#define B_        2
#define S_        1024
#define PAST_     1024
#define KV_       2048     // PAST_ + S_
#define MROWS_    2048     // B_*S_

typedef __bf16 bf16x8  __attribute__((ext_vector_type(8)));
typedef __bf16 bf16x16 __attribute__((ext_vector_type(16)));
typedef float  f32x8   __attribute__((ext_vector_type(8)));

// Load a 16-element bf16 fragment from two 8-element (16-byte) chunks.
__device__ __forceinline__ bf16x16 ld2x8(const __bf16* p0, const __bf16* p1) {
    bf16x8 lo = *(const bf16x8*)p0;
    bf16x8 hi = *(const bf16x8*)p1;
    return __builtin_shufflevector(lo, hi, 0,1,2,3,4,5,6,7,8,9,10,11,12,13,14,15);
}

__device__ __forceinline__ f32x8 wmma_bf16(bf16x16 a, bf16x16 b, f32x8 c) {
    return __builtin_amdgcn_wmma_f32_16x16x32_bf16(
        false, a, false, b, (short)0, c, false, false);
}

// LDS byte offset of a __shared__ object (low 32 bits of the LDS aperture).
__device__ __forceinline__ uint32_t lds_off(const void* p) {
    return (uint32_t)(uintptr_t)p;
}

// Async DMA: 16 bytes global -> LDS per lane, tracked with ASYNCcnt.
__device__ __forceinline__ void async_ld_b128(uint32_t ldsOff, const __bf16* g) {
    asm volatile("global_load_async_to_lds_b128 %0, %1, off"
                 :: "v"(ldsOff), "v"((unsigned long long)(uintptr_t)g)
                 : "memory");
}
__device__ __forceinline__ void wait_async4() {
    asm volatile("s_wait_asynccnt 0x4" ::: "memory");
}
__device__ __forceinline__ void wait_async0() {
    asm volatile("s_wait_asynccnt 0x0" ::: "memory");
}

// ---------------------------------------------------------------------------
// Conversion kernels
// ---------------------------------------------------------------------------
__global__ void cvt_f32_to_bf16(const float* __restrict__ in,
                                __bf16* __restrict__ out, int n) {
    int i = blockIdx.x * blockDim.x + threadIdx.x;
    if (i < n) out[i] = (__bf16)in[i];
}

// past_k [B,H,PAST,D] fp32 -> Kc [B,H,KV,D] bf16 (rows 0..PAST-1)
__global__ void cvt_past_k(const float* __restrict__ in, __bf16* __restrict__ Kc) {
    int i = blockIdx.x * blockDim.x + threadIdx.x;
    int d  = i & (D_ - 1);
    int p  = (i >> 7) & (PAST_ - 1);
    int bh = i >> 17;
    Kc[((size_t)bh * KV_ + p) * D_ + d] = (__bf16)in[i];
}

// past_v [B,H,PAST,D] fp32 -> Vt [B,H,D,KV] bf16 (cols 0..PAST-1), transposed
__global__ void cvt_past_v(const float* __restrict__ in, __bf16* __restrict__ Vt) {
    int i = blockIdx.x * blockDim.x + threadIdx.x;
    int d  = i & (D_ - 1);
    int p  = (i >> 7) & (PAST_ - 1);
    int bh = i >> 17;
    Vt[((size_t)bh * D_ + d) * KV_ + p] = (__bf16)in[i];
}

// ---------------------------------------------------------------------------
// LDS-tiled GEMM: Y[M,N] = A[M,K] * W[N,K]^T + bias, bf16 in, f32 accum.
// Block (8 waves) -> 128x128 tile; wave -> 64x32 (4x2 WMMA accumulators).
// Per 32-k step, A(128x32) and B(128x32) streamed straight into LDS via
// async loads, double-buffered so the next tile's DMA overlaps the WMMAs.
// mode 0: Q->Qb[B,H,S,D]  1: K->Kc[B,H,KV,D](+PAST)  2: fp32 out  3: V->Vt^T
// ---------------------------------------------------------------------------
#define TSTR 40   // LDS row stride in elements (32 data + 8 pad)

__global__ void __launch_bounds__(256)
gemm_wmma_bf16(const __bf16* __restrict__ A, const __bf16* __restrict__ W,
               const float* __restrict__ bias,
               __bf16* __restrict__ outB, float* __restrict__ outF,
               int M, int K, int N, int mode) {
    __shared__ __bf16 Alds[2][128 * TSTR];
    __shared__ __bf16 Blds[2][128 * TSTR];

    const int t    = threadIdx.x;
    const int lane = t & 31;
    const int w    = t >> 5;
    const int half = lane >> 4;
    const int nl   = lane & 15;
    const int wm   = w >> 2;          // wave row (0..1) -> 64 rows
    const int wn   = w & 3;           // wave col (0..3) -> 32 cols

    const int mBase = blockIdx.y * 128;
    const int nBase = blockIdx.x * 128;

    // Cooperative staging: each thread DMAs 32B of A and 32B of B per tile.
    const int srow = t >> 1;          // 0..127
    const int sseg = t & 1;           // 16-elem segment
    const __bf16* agp = A + (size_t)(mBase + srow) * K + sseg * 16;
    const __bf16* bgp = W + (size_t)(nBase + srow) * K + sseg * 16;
    uint32_t aoff[2], boff[2];
    aoff[0] = lds_off(&Alds[0][srow * TSTR + sseg * 16]);
    aoff[1] = lds_off(&Alds[1][srow * TSTR + sseg * 16]);
    boff[0] = lds_off(&Blds[0][srow * TSTR + sseg * 16]);
    boff[1] = lds_off(&Blds[1][srow * TSTR + sseg * 16]);

    auto issue_tile = [&](int buf, int k0) {
        async_ld_b128(aoff[buf],      agp + k0);
        async_ld_b128(aoff[buf] + 16, agp + k0 + 8);
        async_ld_b128(boff[buf],      bgp + k0);
        async_ld_b128(boff[buf] + 16, bgp + k0 + 8);
    };

    f32x8 acc[4][2];
#pragma unroll
    for (int mi = 0; mi < 4; ++mi)
#pragma unroll
        for (int ni = 0; ni < 2; ++ni) acc[mi][ni] = (f32x8){0,0,0,0,0,0,0,0};

    issue_tile(0, 0);                 // prologue DMA of tile 0

    int buf = 0;
    for (int k0 = 0; k0 < K; k0 += 32, buf ^= 1) {
        const bool more = (k0 + 32 < K);
        if (more) issue_tile(buf ^ 1, k0 + 32);   // next tile DMA in flight
        if (more) wait_async4();                  // current tile's 4 done
        else      wait_async0();
        __syncthreads();                          // tile visible to all waves

        // Fragments from LDS, 8 WMMAs per wave.
        bf16x16 af[4], bfr[2];
#pragma unroll
        for (int mi = 0; mi < 4; ++mi) {
            const __bf16* p = &Alds[buf][(wm * 64 + mi * 16 + nl) * TSTR + half * 8];
            af[mi] = ld2x8(p, p + 16);
        }
#pragma unroll
        for (int ni = 0; ni < 2; ++ni) {
            const __bf16* p = &Blds[buf][(wn * 32 + ni * 16 + nl) * TSTR + half * 8];
            bfr[ni] = ld2x8(p, p + 16);
        }
#pragma unroll
        for (int mi = 0; mi < 4; ++mi)
#pragma unroll
            for (int ni = 0; ni < 2; ++ni)
                acc[mi][ni] = wmma_bf16(af[mi], bfr[ni], acc[mi][ni]);
        __syncthreads();              // before next DMA overwrites buf^1... (2 tiles later)
    }

    // Epilogue: lane holds col n=nl, rows half*8+v within each 16x16 tile.
#pragma unroll
    for (int mi = 0; mi < 4; ++mi) {
#pragma unroll
        for (int ni = 0; ni < 2; ++ni) {
            const int n = nBase + wn * 32 + ni * 16 + nl;
            const float bias_v = bias ? bias[n] : 0.0f;
#pragma unroll
            for (int v = 0; v < 8; ++v) {
                const int m = mBase + wm * 64 + mi * 16 + half * 8 + v;
                const float val = acc[mi][ni][v] + bias_v;
                if (mode == 2) {
                    outF[(size_t)m * N + n] = val;
                } else {
                    const int b = m >> 10;           // / S_
                    const int s = m & (S_ - 1);
                    const int h = n >> 7;            // / D_
                    const int d = n & (D_ - 1);
                    const size_t bh = (size_t)b * HEADS_ + h;
                    if (mode == 0)
                        outB[(bh * S_ + s) * D_ + d] = (__bf16)val;
                    else if (mode == 1)
                        outB[(bh * KV_ + (PAST_ + s)) * D_ + d] = (__bf16)val;
                    else // mode 3: transposed V cache
                        outB[(bh * D_ + d) * KV_ + (PAST_ + s)] = (__bf16)val;
                }
            }
        }
    }
}

// ---------------------------------------------------------------------------
// Flash attention, block-cooperative: one block = one (b,h) with 8 q-tiles
// (128 q rows). K(32x128) and Vt(128x32) tiles async-DMAed into
// double-buffered LDS, shared by all 8 waves.
// ---------------------------------------------------------------------------
#define KSTR 136   // K tile LDS row stride (128 + 8 pad)
#define VSTR 40    // V tile LDS row stride (32 + 8 pad)

__global__ void __launch_bounds__(256)
attn_wmma_bf16(const __bf16* __restrict__ Q,   // [B,H,S,D]
               const __bf16* __restrict__ Kc,  // [B,H,KV,D]
               const __bf16* __restrict__ Vt,  // [B,H,D,KV]
               __bf16* __restrict__ Mid) {     // [B,S,HIDDEN]
    __shared__ __bf16 Klds[2][32 * KSTR];      // kv-major: [kv 0..31][d 0..127]
    __shared__ __bf16 Vlds[2][128 * VSTR];     // d-major : [d 0..127][kv 0..31]
    __shared__ __bf16 Plds[8][16 * 32];        // per-wave P tile

    const int t    = threadIdx.x;
    const int lane = t & 31;
    const int w    = t >> 5;
    const int half = lane >> 4;
    const int nl   = lane & 15;

    const int qblk = blockIdx.x & 7;           // S_/128 = 8
    const int bh   = blockIdx.x >> 3;          // 0..31
    const int h    = bh & (HEADS_ - 1);
    const int b    = bh >> 4;
    const int q0   = qblk * 128 + w * 16;      // this wave's 16 q rows

    // Staging coords (32B of K and 32B of V per thread per tile).
    const __bf16* kgp = Kc + ((size_t)bh * KV_ + (t >> 3)) * D_ + (t & 7) * 16;
    const __bf16* vgp = Vt + ((size_t)bh * D_ + (t >> 1)) * KV_ + (t & 1) * 16;
    uint32_t koff[2], voff[2];
    koff[0] = lds_off(&Klds[0][(t >> 3) * KSTR + (t & 7) * 16]);
    koff[1] = lds_off(&Klds[1][(t >> 3) * KSTR + (t & 7) * 16]);
    voff[0] = lds_off(&Vlds[0][(t >> 1) * VSTR + (t & 1) * 16]);
    voff[1] = lds_off(&Vlds[1][(t >> 1) * VSTR + (t & 1) * 16]);

    auto issue_tile = [&](int buf, int kv0) {
        const __bf16* kg = kgp + (size_t)kv0 * D_;
        const __bf16* vg = vgp + kv0;
        async_ld_b128(koff[buf],      kg);
        async_ld_b128(koff[buf] + 16, kg + 8);
        async_ld_b128(voff[buf],      vg);
        async_ld_b128(voff[buf] + 16, vg + 8);
    };

    // Q fragments: 16 rows x 128 k, 4 chunks of 32 (kept in registers).
    bf16x16 qa[4];
    const __bf16* qrow = Q + ((size_t)bh * S_ + q0 + nl) * D_;
#pragma unroll
    for (int c = 0; c < 4; ++c) {
        const __bf16* p = qrow + c * 32 + half * 8;
        qa[c] = ld2x8(p, p + 16);
    }

    f32x8 acc[8];
#pragma unroll
    for (int dc = 0; dc < 8; ++dc) acc[dc] = (f32x8){0,0,0,0,0,0,0,0};
    float mrow[8], lrow[8];
#pragma unroll
    for (int v = 0; v < 8; ++v) { mrow[v] = -1e30f; lrow[v] = 0.0f; }

    const float scale = 0.08838834764831845f;   // 1/sqrt(128)

    issue_tile(0, 0);                           // prologue DMA of kv tile 0

    int buf = 0;
    for (int kv0 = 0; kv0 < KV_; kv0 += 32, buf ^= 1) {
        const bool more = (kv0 + 32 < KV_);
        if (more) issue_tile(buf ^ 1, kv0 + 32);
        if (more) wait_async4();
        else      wait_async0();
        __syncthreads();

        // ---- scores: two 16x16 tiles from LDS K fragments
        float s0[8], s1[8];
#pragma unroll
        for (int sub = 0; sub < 2; ++sub) {
            f32x8 sc = (f32x8){0,0,0,0,0,0,0,0};
            const __bf16* kr = &Klds[buf][(sub * 16 + nl) * KSTR + half * 8];
#pragma unroll
            for (int c = 0; c < 4; ++c) {
                bf16x16 kf = ld2x8(kr + c * 32, kr + c * 32 + 16);
                sc = wmma_bf16(qa[c], kf, sc);
            }
            const int kvpos = kv0 + sub * 16 + nl;           // lane's column
            const bool pad  = (b == 1) && (kvpos >= KV_ - 128);
#pragma unroll
            for (int v = 0; v < 8; ++v) {
                const int qpos = PAST_ + q0 + half * 8 + v;
                float sv = sc[v] * scale;
                if (kvpos > qpos) sv += -1e9f;               // causal
                if (pad)          sv += -1e9f;               // padding
                if (sub) s1[v] = sv; else s0[v] = sv;
            }
        }

        // ---- online softmax (row reductions across 16-lane halves)
#pragma unroll
        for (int v = 0; v < 8; ++v) {
            float tm = fmaxf(s0[v], s1[v]);
            tm = fmaxf(tm, __shfl_xor(tm, 1, 32));
            tm = fmaxf(tm, __shfl_xor(tm, 2, 32));
            tm = fmaxf(tm, __shfl_xor(tm, 4, 32));
            tm = fmaxf(tm, __shfl_xor(tm, 8, 32));
            const float mn   = fmaxf(mrow[v], tm);
            const float corr = __expf(mrow[v] - mn);
            const float p0   = __expf(s0[v] - mn);
            const float p1   = __expf(s1[v] - mn);
            float rs = p0 + p1;
            rs += __shfl_xor(rs, 1, 32);
            rs += __shfl_xor(rs, 2, 32);
            rs += __shfl_xor(rs, 4, 32);
            rs += __shfl_xor(rs, 8, 32);
            lrow[v] = lrow[v] * corr + rs;
            mrow[v] = mn;
#pragma unroll
            for (int dc = 0; dc < 8; ++dc) acc[dc][v] *= corr;
            const int m = half * 8 + v;
            Plds[w][m * 32 + nl]      = (__bf16)p0;
            Plds[w][m * 32 + 16 + nl] = (__bf16)p1;
        }
        __syncthreads();

        // ---- O += P x V from LDS (P as A-fragment, Vt rows as B-fragments)
        const __bf16* pp = &Plds[w][nl * 32 + half * 8];
        bf16x16 pa = ld2x8(pp, pp + 16);
#pragma unroll
        for (int dc = 0; dc < 8; ++dc) {
            const __bf16* vr = &Vlds[buf][(dc * 16 + nl) * VSTR + half * 8];
            bf16x16 vf = ld2x8(vr, vr + 16);
            acc[dc] = wmma_bf16(pa, vf, acc[dc]);
        }
        __syncthreads();
    }

    // Normalize and write Mid [B,S,HIDDEN] bf16.
#pragma unroll
    for (int v = 0; v < 8; ++v) {
        const float inv = 1.0f / lrow[v];
        const int m = half * 8 + v;
        const size_t row = ((size_t)b * S_ + q0 + m) * HIDDEN_ + (size_t)h * D_;
#pragma unroll
        for (int dc = 0; dc < 8; ++dc)
            Mid[row + dc * 16 + nl] = (__bf16)(acc[dc][v] * inv);
    }
}

// ---------------------------------------------------------------------------
// Host-side launch
// Inputs: 0:X 1:past_k 2:past_v 3:causal_mask 4:pad_mask
//         5:Wq 6:bq 7:Wk 8:bk 9:Wv 10:bv 11:Wo 12:bo
// ---------------------------------------------------------------------------
extern "C" void kernel_launch(void* const* d_in, const int* in_sizes, int n_in,
                              void* d_out, int out_size, void* d_ws, size_t ws_size,
                              hipStream_t stream) {
    const float* X      = (const float*)d_in[0];
    const float* past_k = (const float*)d_in[1];
    const float* past_v = (const float*)d_in[2];
    const float* Wq = (const float*)d_in[5];
    const float* bq = (const float*)d_in[6];
    const float* Wk = (const float*)d_in[7];
    const float* bk = (const float*)d_in[8];
    const float* Wv = (const float*)d_in[9];
    const float* bv = (const float*)d_in[10];
    const float* Wo = (const float*)d_in[11];
    const float* bo = (const float*)d_in[12];
    float* out = (float*)d_out;

    const size_t NELEM = (size_t)MROWS_ * HIDDEN_;   // 2048*2048
    __bf16* Xb  = (__bf16*)d_ws;
    __bf16* Wqb = Xb  + NELEM;
    __bf16* Wkb = Wqb + NELEM;
    __bf16* Wvb = Wkb + NELEM;
    __bf16* Wob = Wvb + NELEM;
    __bf16* Qb  = Wob + NELEM;           // [B,H,S,D]
    __bf16* Kc  = Qb  + NELEM;           // [B,H,KV,D]  (2*NELEM)
    __bf16* Vt  = Kc  + 2 * NELEM;       // [B,H,D,KV]  (2*NELEM)
    __bf16* Mid = Vt  + 2 * NELEM;       // [B,S,HIDDEN]
    (void)ws_size; (void)n_in; (void)in_sizes; (void)out_size;

    const int n = (int)NELEM;
    const dim3 cblk(256), cgrd(n / 256);
    cvt_f32_to_bf16<<<cgrd, cblk, 0, stream>>>(X,  Xb,  n);
    cvt_f32_to_bf16<<<cgrd, cblk, 0, stream>>>(Wq, Wqb, n);
    cvt_f32_to_bf16<<<cgrd, cblk, 0, stream>>>(Wk, Wkb, n);
    cvt_f32_to_bf16<<<cgrd, cblk, 0, stream>>>(Wv, Wvb, n);
    cvt_f32_to_bf16<<<cgrd, cblk, 0, stream>>>(Wo, Wob, n);
    cvt_past_k<<<cgrd, cblk, 0, stream>>>(past_k, Kc);
    cvt_past_v<<<cgrd, cblk, 0, stream>>>(past_v, Vt);

    // Projections: block = 128x128 tile, grid (N/128, M/128)
    const dim3 gblk(256), ggrd(HIDDEN_ / 128, MROWS_ / 128);
    gemm_wmma_bf16<<<ggrd, gblk, 0, stream>>>(Xb, Wqb, bq, Qb, nullptr,
                                              MROWS_, HIDDEN_, HIDDEN_, 0);
    gemm_wmma_bf16<<<ggrd, gblk, 0, stream>>>(Xb, Wkb, bk, Kc, nullptr,
                                              MROWS_, HIDDEN_, HIDDEN_, 1);
    gemm_wmma_bf16<<<ggrd, gblk, 0, stream>>>(Xb, Wvb, bv, Vt, nullptr,
                                              MROWS_, HIDDEN_, HIDDEN_, 3);

    // Attention: 256 blocks = 32 (b,h) x 8 q-blocks of 128 rows.
    attn_wmma_bf16<<<dim3(256), dim3(256), 0, stream>>>(Qb, Kc, Vt, Mid);

    // Output projection -> fp32 d_out.
    gemm_wmma_bf16<<<ggrd, gblk, 0, stream>>>(Mid, Wob, bo, nullptr, out,
                                              MROWS_, HIDDEN_, HIDDEN_, 2);
}
